// ConditionalMamba_90898687853228
// MI455X (gfx1250) — matrix operand order repaired and checked by reading, not compiled, not run on hardware
//
#include <hip/hip_runtime.h>
#include <hip/hip_bf16.h>
#include <math.h>

// ---------------------------------------------------------------------------
// ConditionalMamba forward for MI455X (gfx1250, wave32, WMMA).
// Matrix work: V_WMMA_F32_16X16X32_BF16 with split-bf16 (hi+lo) operands and
// fp32 accumulate => ~fp32 accuracy at 3 WMMAs per 16x16x32 tile-chunk.
// All operands are pre-split ONCE into persistent bf16 hi/lo arrays, so the
// WMMA inner loops are pure b128 loads + wmma (no per-tile conversion VALU).
// ---------------------------------------------------------------------------

typedef __bf16        v16bf __attribute__((ext_vector_type(16)));
typedef float         v8f   __attribute__((ext_vector_type(8)));
typedef unsigned int  v4u   __attribute__((ext_vector_type(4)));
typedef unsigned short u16;

#define WMMA_BF16(a, b, c) \
  __builtin_amdgcn_wmma_f32_16x16x32_bf16(false, (a), false, (b), (short)0, (c), false, false)

union BFV { v16bf v; v4u q[2]; u16 u[16]; };

// fp32 -> (bf16 hi, bf16 lo) split, RNE rounding, via bit ops.
__device__ __forceinline__ void bsplit(float x, u16& h, u16& l) {
  unsigned int u0 = __float_as_uint(x);
  unsigned int r0 = u0 + 0x7FFFu + ((u0 >> 16) & 1u);
  u16 hs = (u16)(r0 >> 16);
  float hf = __uint_as_float(((unsigned int)hs) << 16);
  unsigned int u1 = __float_as_uint(x - hf);
  unsigned int r1 = u1 + 0x7FFFu + ((u1 >> 16) & 1u);
  h = hs;
  l = (u16)(r1 >> 16);
}

__device__ __forceinline__ float leakyf(float x)    { return x >= 0.f ? x : 0.01f * x; }
__device__ __forceinline__ float sigm(float x)      { return 1.f / (1.f + __expf(-x)); }
__device__ __forceinline__ float softplusf_(float x){ return x > 20.f ? x : log1pf(__expf(x)); }

// ---------------------------------------------------------------------------
// Generic one-shot fp32 -> bf16 hi/lo split.
// ---------------------------------------------------------------------------
__global__ __launch_bounds__(256) void split_f32(
    const float* __restrict__ src, u16* __restrict__ hi, u16* __restrict__ lo, int n)
{
  const int i = blockIdx.x * 256 + threadIdx.x;
  if (i < n) {
    u16 h, l;
    bsplit(src[i], h, l);
    hi[i] = h;
    lo[i] = l;
  }
}

// ---------------------------------------------------------------------------
// One-shot conv-weight repack+split: OIHW -> [tap][cout][cin] bf16 hi/lo.
// ---------------------------------------------------------------------------
__global__ __launch_bounds__(256) void repack_conv_w_split(
    const float* __restrict__ w, u16* __restrict__ wh, u16* __restrict__ wl)
{
  const int i    = blockIdx.x * 256 + threadIdx.x;  // 9*128*128 = 147456
  const int cin  = i & 127;
  const int cout = (i >> 7) & 127;
  const int tap  = i >> 14;
  u16 h, l;
  bsplit(w[((cout * 128 + cin) * 9) + tap], h, l);
  wh[i] = h;
  wl[i] = l;
}

// ---------------------------------------------------------------------------
// 3x3 SAME conv, 128->128 ch, + bias + leaky. Implicit GEMM, split-bf16 WMMA.
// In/out activations are bf16 hi/lo NCHW arrays. grid=(64,B), block=256
// (8 waves, wave -> 16 couts). wmma-M = cout (A = repacked weights),
// wmma-N = pixel (B = LDS halo tiles). Inner loop: 8 b128 loads + 3 WMMAs.
// ---------------------------------------------------------------------------
#define CSTRB 136  // cin stride (u16) in LDS tiles; %8==0 keeps b128 alignment

__global__ __launch_bounds__(256) void conv3x3_leaky_wmma(
    const u16* __restrict__ in_h, const u16* __restrict__ in_l,
    const u16* __restrict__ wrep_h, const u16* __restrict__ wrep_l,
    const float* __restrict__ bias,
    u16* __restrict__ out_h, u16* __restrict__ out_l)
{
  __shared__ u16 th[3 * 18 * CSTRB];
  __shared__ u16 tl[3 * 18 * CSTRB];
  const int tid = threadIdx.x;
  const int b   = blockIdx.y;
  const int ty  = blockIdx.x >> 1;
  const int x0  = (blockIdx.x & 1) << 4;

  // Cooperative halo load (zero-padded SAME boundary): 3*18*128 elems x2.
  for (int e = tid; e < 3 * 18 * 128; e += 256) {
    const int cin = e / 54;
    const int rem = e % 54;
    const int r   = rem / 18;
    const int col = rem % 18;
    const int gy  = ty + r - 1;
    const int gx  = x0 + col - 1;
    u16 vh = 0, vl = 0;
    if (gy >= 0 && gy < 32 && gx >= 0 && gx < 32) {
      const int gi = ((b * 128 + cin) << 10) + (gy << 5) + gx;
      vh = in_h[gi];
      vl = in_l[gi];
    }
    const int li = (r * 18 + col) * CSTRB + cin;
    th[li] = vh;
    tl[li] = vl;
  }
  __syncthreads();

  const int wv   = tid >> 5;
  const int lane = tid & 31;
  const int lm   = lane & 15;
  const int kh8  = (lane >= 16) ? 8 : 0;    // unused: A fragment also uses bo
  const int bo   = (lane >= 16) ? 16 : 0;   // B-fragment K offset
  const int cb0  = wv << 4;                 // cout base for this wave

  v8f acc = {0.f, 0.f, 0.f, 0.f, 0.f, 0.f, 0.f, 0.f};

  for (int ky = 0; ky < 3; ++ky) {
    for (int kx = 0; kx < 3; ++kx) {
      const int tap  = ky * 3 + kx;
      const u16* tph = &th[(ky * 18 + lm + kx) * CSTRB];  // pixel = lm
      const u16* tpl = &tl[(ky * 18 + lm + kx) * CSTRB];
      const u16* aph = wrep_h + (((size_t)tap * 128 + cb0 + lm) << 7);  // cout = lm
      const u16* apl = wrep_l + (((size_t)tap * 128 + cb0 + lm) << 7);
      for (int cb = 0; cb < 128; cb += 32) {
        BFV ah, al, bh, bl;
        ah.q[0] = *reinterpret_cast<const v4u*>(aph + cb + bo);
        ah.q[1] = *reinterpret_cast<const v4u*>(aph + cb + bo + 8);
        al.q[0] = *reinterpret_cast<const v4u*>(apl + cb + bo);
        al.q[1] = *reinterpret_cast<const v4u*>(apl + cb + bo + 8);
        bh.q[0] = *reinterpret_cast<const v4u*>(tph + cb + bo);
        bh.q[1] = *reinterpret_cast<const v4u*>(tph + cb + bo + 8);
        bl.q[0] = *reinterpret_cast<const v4u*>(tpl + cb + bo);
        bl.q[1] = *reinterpret_cast<const v4u*>(tpl + cb + bo + 8);
        acc = WMMA_BF16(ah.v, bh.v, acc);
        acc = WMMA_BF16(ah.v, bl.v, acc);
        acc = WMMA_BF16(al.v, bh.v, acc);
      }
    }
  }

  const int hiOff = (lane >= 16) ? 8 : 0;
  #pragma unroll
  for (int r = 0; r < 8; ++r) {
    const int cout = cb0 + r + hiOff;
    const float v  = leakyf(acc[r] + bias[cout]);
    u16 h, l;
    bsplit(v, h, l);
    const int oi = ((b * 128 + cout) << 10) + (ty << 5) + x0 + lm;
    out_h[oi] = h;
    out_l[oi] = l;
  }
}

// ---------------------------------------------------------------------------
// Generic split-bf16 WMMA GEMM: Y[m,n] = sum_k X[m,k] * W[n,k].
// X/W given as bf16 hi/lo arrays. Output fp32, optionally also bf16 hi/lo.
// Fast path: pure b128 loads + 3 WMMAs; guarded path for Kd%32!=0 (dt_proj).
// grid = (M/16, ceil(ntiles/8)); block = 256; one n-tile per wave.
// ---------------------------------------------------------------------------
__global__ __launch_bounds__(256) void gemm_rowout_wmma(
    const u16* __restrict__ Xh, const u16* __restrict__ Xl,
    const u16* __restrict__ Wh, const u16* __restrict__ Wl,
    float* __restrict__ Yf, u16* __restrict__ Yh, u16* __restrict__ Yl,
    int M, int Kd, int N, int ldx, int ldy, int emitHL)
{
  const int tid  = threadIdx.x;
  const int wv   = tid >> 5;
  const int lane = tid & 31;
  const int lm   = lane & 15;
  const int kh8  = (lane >= 16) ? 8 : 0;
  const int bo   = (lane >= 16) ? 16 : 0;

  const int ntiles = (N + 15) >> 4;
  const int ntile  = blockIdx.y * 8 + wv;
  if (ntile >= ntiles) return;  // wave-uniform exit

  const int m0 = blockIdx.x << 4;
  const int n0 = ntile << 4;
  const int m  = m0 + lm;
  const int n  = n0 + lm;

  v8f acc = {0.f, 0.f, 0.f, 0.f, 0.f, 0.f, 0.f, 0.f};
  const u16* xh = Xh + (size_t)m * ldx;
  const u16* xl = Xl + (size_t)m * ldx;
  const u16* wh = Wh + (size_t)n * Kd;
  const u16* wl = Wl + (size_t)n * Kd;
  const bool nok = (n < N);
  const v4u z4 = {0u, 0u, 0u, 0u};

  for (int k = 0; k < Kd; k += 32) {
    BFV ah, al, bh, bl;
    if (k + 32 <= Kd) {                    // uniform fast path: b128 loads only
      ah.q[0] = *reinterpret_cast<const v4u*>(xh + k + bo);
      ah.q[1] = *reinterpret_cast<const v4u*>(xh + k + bo + 8);
      al.q[0] = *reinterpret_cast<const v4u*>(xl + k + bo);
      al.q[1] = *reinterpret_cast<const v4u*>(xl + k + bo + 8);
      if (nok) {
        bh.q[0] = *reinterpret_cast<const v4u*>(wh + k + bo);
        bh.q[1] = *reinterpret_cast<const v4u*>(wh + k + bo + 8);
        bl.q[0] = *reinterpret_cast<const v4u*>(wl + k + bo);
        bl.q[1] = *reinterpret_cast<const v4u*>(wl + k + bo + 8);
      } else {
        bh.q[0] = z4; bh.q[1] = z4; bl.q[0] = z4; bl.q[1] = z4;
      }
    } else {                               // K tail (dt_proj Kd=8): guarded
      #pragma unroll
      for (int j = 0; j < 8; ++j) {
        const int k0 = k + bo + j;
        const int k1 = k + bo + 8 + j;
        ah.u[j]     = (k0 < Kd) ? xh[k0] : 0;
        al.u[j]     = (k0 < Kd) ? xl[k0] : 0;
        ah.u[8 + j] = (k1 < Kd) ? xh[k1] : 0;
        al.u[8 + j] = (k1 < Kd) ? xl[k1] : 0;
      }
      #pragma unroll
      for (int j = 0; j < 16; ++j) {
        const int kk = k + bo + j;
        const bool ok = nok && (kk < Kd);
        bh.u[j] = ok ? wh[kk] : 0;
        bl.u[j] = ok ? wl[kk] : 0;
      }
    }
    acc = WMMA_BF16(ah.v, bh.v, acc);
    acc = WMMA_BF16(ah.v, bl.v, acc);
    acc = WMMA_BF16(al.v, bh.v, acc);
  }

  const int hiOff = (lane >= 16) ? 8 : 0;
  #pragma unroll
  for (int r = 0; r < 8; ++r) {
    const int mm = m0 + r + hiOff;
    if (nok) {
      const size_t oi = (size_t)mm * ldy + n;
      Yf[oi] = acc[r];
      if (emitHL) {
        u16 h, l;
        bsplit(acc[r], h, l);
        Yh[oi] = h;
        Yl[oi] = l;
      }
    }
  }
}

// ---------------------------------------------------------------------------
// out_proj: Out[b][cout][l] = sum_d Yin[b][l][d] * Wo[cout][d]; fp32 NC(HW).
// wmma-M = cout (A = Wo hi/lo), wmma-N = l (B = y hi/lo); K = 256.
// ---------------------------------------------------------------------------
__global__ __launch_bounds__(256) void out_proj_wmma(
    const u16* __restrict__ Yh, const u16* __restrict__ Yl,
    const u16* __restrict__ Wh, const u16* __restrict__ Wl,
    float* __restrict__ Out)
{
  const int tid  = threadIdx.x;
  const int wv   = tid >> 5;
  const int lane = tid & 31;
  const int lm   = lane & 15;
  const int kh8  = (lane >= 16) ? 8 : 0;
  const int bo   = (lane >= 16) ? 16 : 0;
  const int b    = blockIdx.y;
  const int l0   = blockIdx.x << 4;
  const int cb0  = wv << 4;

  v8f acc = {0.f, 0.f, 0.f, 0.f, 0.f, 0.f, 0.f, 0.f};
  const u16* wh = Wh + (size_t)(cb0 + lm) * 256;
  const u16* wl = Wl + (size_t)(cb0 + lm) * 256;
  const u16* yh = Yh + (size_t)((b << 10) + l0 + lm) * 256;
  const u16* yl = Yl + (size_t)((b << 10) + l0 + lm) * 256;

  for (int k = 0; k < 256; k += 32) {
    BFV ah, al, bh, bl;
    ah.q[0] = *reinterpret_cast<const v4u*>(wh + k + bo);
    ah.q[1] = *reinterpret_cast<const v4u*>(wh + k + bo + 8);
    al.q[0] = *reinterpret_cast<const v4u*>(wl + k + bo);
    al.q[1] = *reinterpret_cast<const v4u*>(wl + k + bo + 8);
    bh.q[0] = *reinterpret_cast<const v4u*>(yh + k + bo);
    bh.q[1] = *reinterpret_cast<const v4u*>(yh + k + bo + 8);
    bl.q[0] = *reinterpret_cast<const v4u*>(yl + k + bo);
    bl.q[1] = *reinterpret_cast<const v4u*>(yl + k + bo + 8);
    acc = WMMA_BF16(ah.v, bh.v, acc);
    acc = WMMA_BF16(ah.v, bl.v, acc);
    acc = WMMA_BF16(al.v, bh.v, acc);
  }

  const int hiOff = (lane >= 16) ? 8 : 0;
  #pragma unroll
  for (int r = 0; r < 8; ++r) {
    const int cout = cb0 + r + hiOff;
    Out[((b * 128 + cout) << 10) + l0 + lm] = acc[r];
  }
}

// ---------------------------------------------------------------------------
// Concat + transpose (hi/lo): xcat[b][l][d] from conv branch outputs.
// ---------------------------------------------------------------------------
__global__ __launch_bounds__(256) void concat_seq_hl(
    const u16* __restrict__ cCh, const u16* __restrict__ cCl,
    const u16* __restrict__ cPh, const u16* __restrict__ cPl,
    u16* __restrict__ xh, u16* __restrict__ xl)
{
  const int i  = blockIdx.x * 256 + threadIdx.x;  // 8*2048*128 total
  const int dd = i & 127;
  const int l  = (i >> 7) & 2047;
  const int b  = i >> 18;
  int si;
  const u16 *sh, *sl;
  if (l < 1024) {
    si = ((b * 128 + dd) << 10) + l;
    sh = cCh; sl = cCl;
  } else {
    si = ((b * 128 + dd) << 10) + (l - 1024);
    sh = cPh; sl = cPl;
  }
  xh[i] = sh[si];
  xl[i] = sl[si];
}

// ---------------------------------------------------------------------------
// Depthwise causal conv1d (K=4) + SiLU: u fp32 + bf16 hi/lo (feeds x_proj).
// ---------------------------------------------------------------------------
__global__ __launch_bounds__(256) void dwconv1d_silu(
    const float* __restrict__ xz, const float* __restrict__ cw,
    const float* __restrict__ cbias, float* __restrict__ u,
    u16* __restrict__ uh, u16* __restrict__ ul)
{
  const int i = blockIdx.x * 256 + threadIdx.x;  // 8*2048*256 total
  const int d = i & 255;
  const int t = (i >> 8) & 2047;
  const int b = i >> 19;
  float acc = cbias[d];
  #pragma unroll
  for (int j = 0; j < 4; ++j) {
    const int tt = t - 3 + j;
    if (tt >= 0)
      acc = fmaf(xz[(size_t)((b * 2048 + tt) << 9) + d], cw[d * 4 + j], acc);
  }
  const float uv = acc * sigm(acc);
  u[i] = uv;
  u16 h, l;
  bsplit(uv, h, l);
  uh[i] = h;
  ul[i] = l;
}

// ---------------------------------------------------------------------------
// Selective scan: block = batch, thread = channel d, 16 states in registers.
// Emits y as bf16 hi/lo (only consumed by out_proj WMMA), last Lp steps only.
// ---------------------------------------------------------------------------
__global__ __launch_bounds__(256) void ssm_scan(
    const float* __restrict__ dtraw, const float* __restrict__ dtb,
    const float* __restrict__ A_log, const float* __restrict__ u,
    const float* __restrict__ xdbl, const float* __restrict__ xz,
    const float* __restrict__ Dp, u16* __restrict__ yh, u16* __restrict__ yl)
{
  __shared__ float sBC[32];  // [0..15]=B_t, [16..31]=C_t
  const int b = blockIdx.x;
  const int d = threadIdx.x;

  float h[16], Av[16];
  #pragma unroll
  for (int n = 0; n < 16; ++n) {
    h[n]  = 0.f;
    Av[n] = -__expf(A_log[d * 16 + n]);  // A = -exp(A_log)
  }
  const float bdt = dtb[d];
  const float Dv  = Dp[d];

  for (int t = 0; t < 2048; ++t) {
    __syncthreads();
    if (d < 32) sBC[d] = xdbl[(size_t)(b * 2048 + t) * 40 + 8 + d];
    __syncthreads();

    const int row   = b * 2048 + t;
    const float dtv = softplusf_(dtraw[(size_t)row * 256 + d] + bdt);
    const float uu  = u[(size_t)row * 256 + d];
    float acc = 0.f;
    #pragma unroll
    for (int n = 0; n < 16; ++n) {
      const float dA = __expf(dtv * Av[n]);
      h[n] = fmaf(dA, h[n], dtv * sBC[n] * uu);
      acc  = fmaf(h[n], sBC[16 + n], acc);
    }
    if (t >= 1024) {
      const float zv = xz[(size_t)(row << 9) + 256 + d];
      float yv = fmaf(uu, Dv, acc);
      yv *= zv * sigm(zv);
      u16 hh, ll;
      bsplit(yv, hh, ll);
      const size_t oi = (size_t)((b << 10) + (t - 1024)) * 256 + d;
      yh[oi] = hh;
      yl[oi] = ll;
    }
  }
}

// ---------------------------------------------------------------------------
// Host launcher. Workspace carved in bytes (~125 MB).
// ---------------------------------------------------------------------------
extern "C" void kernel_launch(void* const* d_in, const int* in_sizes, int n_in,
                              void* d_out, int out_size, void* d_ws, size_t ws_size,
                              hipStream_t stream)
{
  (void)in_sizes; (void)n_in; (void)out_size; (void)ws_size;

  const float* primary   = (const float*)d_in[0];
  const float* condx     = (const float*)d_in[1];
  const float* cw1 = (const float*)d_in[2];  const float* cb1 = (const float*)d_in[3];
  const float* cw2 = (const float*)d_in[4];  const float* cb2 = (const float*)d_in[5];
  const float* pw1 = (const float*)d_in[6];  const float* pb1 = (const float*)d_in[7];
  const float* pw2 = (const float*)d_in[8];  const float* pb2 = (const float*)d_in[9];
  const float* in_proj_w  = (const float*)d_in[10];
  const float* conv1d_w   = (const float*)d_in[11];
  const float* conv1d_b   = (const float*)d_in[12];
  const float* x_proj_w   = (const float*)d_in[13];
  const float* dt_proj_w  = (const float*)d_in[14];
  const float* dt_proj_b  = (const float*)d_in[15];
  const float* A_log      = (const float*)d_in[16];
  const float* D_param    = (const float*)d_in[17];
  const float* out_proj_w = (const float*)d_in[18];
  float* out = (float*)d_out;

  char* p = (char*)d_ws;
  auto carve = [&](size_t bytes) -> void* {
    void* r = (void*)p;
    p += (bytes + 255) & ~(size_t)255;
    return r;
  };

  // fp32 buffers
  float* xz    = (float*)carve((size_t)8388608 * 4);  // B*2048*512
  float* ubuf  = (float*)carve((size_t)4194304 * 4);  // B*2048*256
  float* xdbl  = (float*)carve((size_t)655360  * 4);  // B*2048*40
  float* dtraw = (float*)carve((size_t)4194304 * 4);  // B*2048*256

  // bf16 hi/lo activation buffers
  u16* cond_h = (u16*)carve((size_t)1048576 * 2);
  u16* cond_l = (u16*)carve((size_t)1048576 * 2);
  u16* prim_h = (u16*)carve((size_t)1048576 * 2);
  u16* prim_l = (u16*)carve((size_t)1048576 * 2);
  u16* tmp_h  = (u16*)carve((size_t)1048576 * 2);
  u16* tmp_l  = (u16*)carve((size_t)1048576 * 2);
  u16* cC_h   = (u16*)carve((size_t)1048576 * 2);
  u16* cC_l   = (u16*)carve((size_t)1048576 * 2);
  u16* cP_h   = (u16*)carve((size_t)1048576 * 2);
  u16* cP_l   = (u16*)carve((size_t)1048576 * 2);
  u16* xcat_h = (u16*)carve((size_t)2097152 * 2);
  u16* xcat_l = (u16*)carve((size_t)2097152 * 2);
  u16* u_h    = (u16*)carve((size_t)4194304 * 2);
  u16* u_l    = (u16*)carve((size_t)4194304 * 2);
  u16* xdbl_h = (u16*)carve((size_t)655360  * 2);
  u16* xdbl_l = (u16*)carve((size_t)655360  * 2);
  u16* y_h    = (u16*)carve((size_t)2097152 * 2);
  u16* y_l    = (u16*)carve((size_t)2097152 * 2);

  // bf16 hi/lo weight buffers
  u16* wr1_h  = (u16*)carve((size_t)147456 * 2);
  u16* wr1_l  = (u16*)carve((size_t)147456 * 2);
  u16* wr2_h  = (u16*)carve((size_t)147456 * 2);
  u16* wr2_l  = (u16*)carve((size_t)147456 * 2);
  u16* wr3_h  = (u16*)carve((size_t)147456 * 2);
  u16* wr3_l  = (u16*)carve((size_t)147456 * 2);
  u16* wr4_h  = (u16*)carve((size_t)147456 * 2);
  u16* wr4_l  = (u16*)carve((size_t)147456 * 2);
  u16* wip_h  = (u16*)carve((size_t)65536 * 2);
  u16* wip_l  = (u16*)carve((size_t)65536 * 2);
  u16* wxp_h  = (u16*)carve((size_t)10240 * 2);
  u16* wxp_l  = (u16*)carve((size_t)10240 * 2);
  u16* wdt_h  = (u16*)carve((size_t)2048  * 2);
  u16* wdt_l  = (u16*)carve((size_t)2048  * 2);
  u16* wout_h = (u16*)carve((size_t)32768 * 2);
  u16* wout_l = (u16*)carve((size_t)32768 * 2);

  const dim3 blk(256);

  // One-shot splits (inputs + GEMM weights) and conv weight repack+split.
  split_f32<<<dim3(4096), blk, 0, stream>>>(condx,      cond_h, cond_l, 1048576);
  split_f32<<<dim3(4096), blk, 0, stream>>>(primary,    prim_h, prim_l, 1048576);
  split_f32<<<dim3(256),  blk, 0, stream>>>(in_proj_w,  wip_h,  wip_l,  65536);
  split_f32<<<dim3(40),   blk, 0, stream>>>(x_proj_w,   wxp_h,  wxp_l,  10240);
  split_f32<<<dim3(8),    blk, 0, stream>>>(dt_proj_w,  wdt_h,  wdt_l,  2048);
  split_f32<<<dim3(128),  blk, 0, stream>>>(out_proj_w, wout_h, wout_l, 32768);
  repack_conv_w_split<<<dim3(576), blk, 0, stream>>>(cw1, wr1_h, wr1_l);
  repack_conv_w_split<<<dim3(576), blk, 0, stream>>>(cw2, wr2_h, wr2_l);
  repack_conv_w_split<<<dim3(576), blk, 0, stream>>>(pw1, wr3_h, wr3_l);
  repack_conv_w_split<<<dim3(576), blk, 0, stream>>>(pw2, wr4_h, wr4_l);

  // Two conv stacks (tmp reused between branches).
  conv3x3_leaky_wmma<<<dim3(64, 8), blk, 0, stream>>>(cond_h, cond_l, wr1_h, wr1_l, cb1, tmp_h, tmp_l);
  conv3x3_leaky_wmma<<<dim3(64, 8), blk, 0, stream>>>(tmp_h,  tmp_l,  wr2_h, wr2_l, cb2, cC_h,  cC_l);
  conv3x3_leaky_wmma<<<dim3(64, 8), blk, 0, stream>>>(prim_h, prim_l, wr3_h, wr3_l, pb1, tmp_h, tmp_l);
  conv3x3_leaky_wmma<<<dim3(64, 8), blk, 0, stream>>>(tmp_h,  tmp_l,  wr4_h, wr4_l, pb2, cP_h,  cP_l);

  // Sequence assembly + projections.
  concat_seq_hl<<<dim3(8192), blk, 0, stream>>>(cC_h, cC_l, cP_h, cP_l, xcat_h, xcat_l);
  // in_proj: M=16384, K=128, N=512 -> xz fp32
  gemm_rowout_wmma<<<dim3(1024, 4), blk, 0, stream>>>(xcat_h, xcat_l, wip_h, wip_l,
      xz, nullptr, nullptr, 16384, 128, 512, 128, 512, 0);
  dwconv1d_silu<<<dim3(16384), blk, 0, stream>>>(xz, conv1d_w, conv1d_b, ubuf, u_h, u_l);
  // x_proj: M=16384, K=256, N=40 -> xdbl fp32 + hi/lo
  gemm_rowout_wmma<<<dim3(1024, 1), blk, 0, stream>>>(u_h, u_l, wxp_h, wxp_l,
      xdbl, xdbl_h, xdbl_l, 16384, 256, 40, 256, 40, 1);
  // dt_proj: M=16384, K=8 (guarded), N=256; X = x_dbl[:, :8] hi/lo, stride 40
  gemm_rowout_wmma<<<dim3(1024, 2), blk, 0, stream>>>(xdbl_h, xdbl_l, wdt_h, wdt_l,
      dtraw, nullptr, nullptr, 16384, 8, 256, 40, 256, 0);

  // Sequential SSM scan (critical path) + gating; emits y hi/lo, last Lp only.
  ssm_scan<<<dim3(8), blk, 0, stream>>>(dtraw, dt_proj_b, A_log, ubuf,
                                        xdbl, xz, D_param, y_h, y_l);

  // out_proj straight into (B, C, H*W) fp32 output layout.
  out_proj_wmma<<<dim3(64, 8), blk, 0, stream>>>(y_h, y_l, wout_h, wout_l, out);
}
